// MultiHeadAttention_7095285973076
// MI455X (gfx1250) — matrix operand order repaired and checked by reading, not compiled, not run on hardware
//
#include <hip/hip_runtime.h>
#include <hip/hip_bf16.h>

// ---------------------------------------------------------------------------
// MultiHeadAttention for MI455X (gfx1250)
//  - all matmuls via v_wmma_f32_16x16x32_bf16 (fp32 accumulate)
//  - global->LDS staging via async-to-LDS (ASYNCcnt) + double buffering
// ---------------------------------------------------------------------------

typedef __bf16 bf16_t;
typedef __attribute__((ext_vector_type(16))) __bf16 v16bf;
typedef __attribute__((ext_vector_type(8)))  __bf16 v8bf;
typedef __attribute__((ext_vector_type(8)))  float  v8f;

#define D_MODEL 4096
#define SEQ     4096
#define N_HEADS 16
#define CHUNK   256   // rows per head-chunk

#define BM 256
#define BN 128
#define BK 32
#define LDS_STRIDE 40   // 32 + 8 halves: rows 16B aligned, banks skewed

union FragAB { v16bf v; v8bf h[2]; };

// Async copy of 16B from global to LDS; tracked by ASYNCcnt.
__device__ __forceinline__ void async_ld_b128(const bf16_t* gptr, bf16_t* lptr) {
    unsigned long long ga = (unsigned long long)(uintptr_t)gptr;
    unsigned la = (unsigned)(uintptr_t)lptr;   // low 32b of generic = LDS offset
    asm volatile("global_load_async_to_lds_b128 %0, %1, off"
                 :: "v"(la), "v"(ga) : "memory");
}
__device__ __forceinline__ void wait_async0() {
    asm volatile("s_wait_asynccnt 0x0" ::: "memory");
}

// ---------------- fp32 -> bf16 conversion ----------------
__global__ void convert_f32_bf16(const float* __restrict__ in,
                                 bf16_t* __restrict__ out, int n) {
    int i = (blockIdx.x * blockDim.x + threadIdx.x) * 4;
    if (i + 3 < n) {
        float4 f = *(const float4*)(in + i);
        out[i + 0] = (bf16_t)f.x;
        out[i + 1] = (bf16_t)f.y;
        out[i + 2] = (bf16_t)f.z;
        out[i + 3] = (bf16_t)f.w;
    }
}

// ---------------- NT GEMM: C[m,n] = sum_k A[m,k]*B[n,k] (+bias[n]) ----------
// Block tile 256x128, BK=32, double-buffered async LDS staging.
// 256 threads = 8 waves; wave (wM=w&3 -> 64 rows, wN=w>>2 -> 64 cols)
// owns a 64x64 patch = 4x4 WMMA tiles (16 WMMAs per K-step).
template <bool BF16_OUT, bool HAS_BIAS>
__global__ __launch_bounds__(256)
void gemm_nt_wmma(const bf16_t* __restrict__ A, const bf16_t* __restrict__ B,
                  const float* __restrict__ bias, void* __restrict__ Cout,
                  int M, int N, int K,
                  long long sA, long long sB, long long sC) {
    __shared__ bf16_t As[2][BM][LDS_STRIDE];   // 2 x 256 x 40 x 2B = 40 KB
    __shared__ bf16_t Bs[2][BN][LDS_STRIDE];   // 2 x 128 x 40 x 2B = 20 KB

    const int tid  = threadIdx.x;
    const int lane = tid & 31;
    const int wave = tid >> 5;
    const int l16  = lane & 15;
    const int hig  = (lane >> 4) & 1;   // upper half-wave?

    const int waveM = wave & 3;   // 4 waves along M: 64 rows each
    const int waveN = wave >> 2;  // 2 waves along N: 64 cols each

    const long long z = blockIdx.z;
    A += z * sA;
    B += z * sB;

    const int m0 = blockIdx.x * BM;
    const int n0 = blockIdx.y * BN;

    const v8f vzero = {};
    v8f acc[4][4];
#pragma unroll
    for (int i = 0; i < 4; ++i)
#pragma unroll
        for (int j = 0; j < 4; ++j) acc[i][j] = vzero;

    // stage one BMxBK A tile + BNxBK B tile into LDS buffer `buf` (async)
    auto stage = [&](int buf, int k0) {
#pragma unroll
        for (int p = 0; p < 4; ++p) {           // A: 256x32 = 1024 x b128
            int id  = tid + p * 256;
            int row = id >> 2;
            int col = (id & 3) << 3;
            async_ld_b128(A + (long long)(m0 + row) * K + k0 + col,
                          &As[buf][row][col]);
        }
#pragma unroll
        for (int p = 0; p < 2; ++p) {           // B: 128x32 = 512 x b128
            int id  = tid + p * 256;
            int row = id >> 2;
            int col = (id & 3) << 3;
            async_ld_b128(B + (long long)(n0 + row) * K + k0 + col,
                          &Bs[buf][row][col]);
        }
    };

    int buf = 0;
    stage(buf, 0);
    wait_async0();
    __syncthreads();

    for (int k0 = 0; k0 < K; k0 += BK) {
        const int nbuf = buf ^ 1;
        if (k0 + BK < K) stage(nbuf, k0 + BK);   // prefetch next tile (async)

        // ---- fragment loads per ISA 7.12.2 lane layouts ----
        FragAB fa[4], fb[4];
#pragma unroll
        for (int i = 0; i < 4; ++i) {
            // A 16x32: lane<16 -> K{0..7,16..23}; lane>=16 -> K{8..15,24..31}
            const bf16_t* p = &As[buf][waveM * 64 + i * 16 + l16][hig ? 8 : 0];
            fa[i].h[0] = *(const v8bf*)p;
            fa[i].h[1] = *(const v8bf*)(p + 16);
        }
#pragma unroll
        for (int j = 0; j < 4; ++j) {
            // B 32x16: lane%16 = column; lane<16 -> K{0..7,16..23}; lane>=16 -> K{8..15,24..31}
            const bf16_t* p = &Bs[buf][waveN * 64 + j * 16 + l16][hig ? 8 : 0];
            fb[j].h[0] = *(const v8bf*)p;
            fb[j].h[1] = *(const v8bf*)(p + 16);
        }

#pragma unroll
        for (int i = 0; i < 4; ++i)
#pragma unroll
            for (int j = 0; j < 4; ++j)
                acc[i][j] = __builtin_amdgcn_wmma_f32_16x16x32_bf16(
                    false, fa[i].v, false, fb[j].v, (short)0, acc[i][j],
                    false, false);

        wait_async0();      // next tile fully landed in LDS
        __syncthreads();    // everyone done reading `buf`
        buf = nbuf;
    }

    // ---- store: C/D layout: VGPR r, lane l -> M=r+(l>=16?8:0), N=l%16 ----
#pragma unroll
    for (int i = 0; i < 4; ++i) {
#pragma unroll
        for (int j = 0; j < 4; ++j) {
            const int gn = n0 + waveN * 64 + j * 16 + l16;
            const float bv = HAS_BIAS ? bias[gn] : 0.f;
#pragma unroll
            for (int r = 0; r < 8; ++r) {
                const int gm = m0 + waveM * 64 + i * 16 + r + (hig ? 8 : 0);
                const float v = acc[i][j][r] + bv;
                if (BF16_OUT)
                    ((bf16_t*)Cout)[z * sC + (long long)gm * N + gn] = (bf16_t)v;
                else
                    ((float*)Cout)[z * sC + (long long)gm * N + gn] = v;
            }
        }
    }
}

// ---------------- per-head transpose: vt[h][d][r] = q[h*256+r][d] ----------
__global__ void transpose_heads(const bf16_t* __restrict__ q,
                                bf16_t* __restrict__ vt) {
    long long idx = (long long)blockIdx.x * blockDim.x + threadIdx.x;
    if (idx < (long long)N_HEADS * D_MODEL * CHUNK) {
        int r = (int)(idx & (CHUNK - 1));
        long long t = idx >> 8;
        int d = (int)(t & (D_MODEL - 1));
        int h = (int)(t >> 12);
        vt[idx] = q[(long long)(h * CHUNK + r) * D_MODEL + d];
    }
}

// ---------------- row softmax with 1/sqrt(D) scale, bf16 out ----------------
__global__ __launch_bounds__(256)
void softmax_rows(const float* __restrict__ s, bf16_t* __restrict__ attn) {
    __shared__ float red[256];
    const int row = blockIdx.x;   // 0..4095 = (head*256 + q-row)
    const int j = threadIdx.x;
    const float scale = 0.015625f;  // 1/sqrt(4096)
    float v = s[(long long)row * 256 + j] * scale;

    red[j] = v;
    __syncthreads();
    for (int off = 128; off > 0; off >>= 1) {
        if (j < off) red[j] = fmaxf(red[j], red[j + off]);
        __syncthreads();
    }
    const float m = red[0];
    __syncthreads();
    const float e = __expf(v - m);
    red[j] = e;
    __syncthreads();
    for (int off = 128; off > 0; off >>= 1) {
        if (j < off) red[j] += red[j + off];
        __syncthreads();
    }
    attn[(long long)row * 256 + j] = (bf16_t)(e / red[0]);
}

// ---------------------------------------------------------------------------
extern "C" void kernel_launch(void* const* d_in, const int* in_sizes, int n_in,
                              void* d_out, int out_size, void* d_ws,
                              size_t ws_size, hipStream_t stream) {
    const float* x  = (const float*)d_in[0];
    const float* Wq = (const float*)d_in[1];
    const float* bq = (const float*)d_in[2];
    const float* Wk = (const float*)d_in[3];
    const float* bk = (const float*)d_in[4];
    const float* Wv = (const float*)d_in[5];
    const float* bv = (const float*)d_in[6];
    const float* Wo = (const float*)d_in[7];
    const float* bo = (const float*)d_in[8];
    float* out = (float*)d_out;

    const size_t MAT_EL = (size_t)SEQ * D_MODEL;          // 16.7M elements
    const size_t MAT_BF = MAT_EL * sizeof(bf16_t);        // 32 MB

    char* ws = (char*)d_ws;
    bf16_t* xb    = (bf16_t*)(ws + 0 * MAT_BF);
    bf16_t* Wqb   = (bf16_t*)(ws + 1 * MAT_BF);
    bf16_t* Wkb   = (bf16_t*)(ws + 2 * MAT_BF);
    bf16_t* Wvb   = (bf16_t*)(ws + 3 * MAT_BF);
    bf16_t* Wob   = (bf16_t*)(ws + 4 * MAT_BF);
    bf16_t* qb    = (bf16_t*)(ws + 5 * MAT_BF);
    bf16_t* kb    = (bf16_t*)(ws + 6 * MAT_BF);
    bf16_t* vb    = (bf16_t*)(ws + 7 * MAT_BF);
    bf16_t* vtb   = (bf16_t*)(ws + 8 * MAT_BF);   // q transposed per head
    bf16_t* catb  = (bf16_t*)(ws + 9 * MAT_BF);   // attention output, concat
    float*  score = (float*) (ws + 10 * MAT_BF);  // 16*256*256 f32 = 4 MB
    bf16_t* attnb = (bf16_t*)(ws + 10 * MAT_BF + (size_t)N_HEADS * CHUNK * CHUNK * 4);

    const dim3 blk(256);

    // 1) fp32 -> bf16 (x + 4 weights)
    {
        const int n = (int)MAT_EL;
        const dim3 g((n / 4 + 255) / 256);
        convert_f32_bf16<<<g, blk, 0, stream>>>(x, xb, n);
        convert_f32_bf16<<<g, blk, 0, stream>>>(Wq, Wqb, n);
        convert_f32_bf16<<<g, blk, 0, stream>>>(Wk, Wkb, n);
        convert_f32_bf16<<<g, blk, 0, stream>>>(Wv, Wvb, n);
        convert_f32_bf16<<<g, blk, 0, stream>>>(Wo, Wob, n);
    }

    // 2) q/k/v projections: [4096,4096] = xb · W^T + bias  (bf16 out)
    {
        const dim3 g(SEQ / BM, D_MODEL / BN, 1);
        gemm_nt_wmma<true, true><<<g, blk, 0, stream>>>(
            xb, Wqb, bq, qb, SEQ, D_MODEL, D_MODEL, 0, 0, 0);
        gemm_nt_wmma<true, true><<<g, blk, 0, stream>>>(
            xb, Wkb, bk, kb, SEQ, D_MODEL, D_MODEL, 0, 0, 0);
        gemm_nt_wmma<true, true><<<g, blk, 0, stream>>>(
            xb, Wvb, bv, vb, SEQ, D_MODEL, D_MODEL, 0, 0, 0);
    }

    // 3) transpose q head-chunks: vtb[h][d][r] = qb[h*256+r][d]
    {
        const long long n = (long long)N_HEADS * D_MODEL * CHUNK;
        transpose_heads<<<dim3((unsigned)((n + 255) / 256)), blk, 0, stream>>>(qb, vtb);
    }

    // 4) scores[h] = Qs·Ks^T  with Qs = split(v), Ks = split(k)  (fp32 out)
    {
        const dim3 g(CHUNK / BM, CHUNK / BN, N_HEADS);   // (1, 2, 16)
        gemm_nt_wmma<false, false><<<g, blk, 0, stream>>>(
            vb, kb, nullptr, score, CHUNK, CHUNK, D_MODEL,
            (long long)CHUNK * D_MODEL, (long long)CHUNK * D_MODEL,
            (long long)CHUNK * CHUNK);
    }

    // 5) softmax over last dim, scale 1/64, bf16 out
    softmax_rows<<<dim3(N_HEADS * CHUNK), blk, 0, stream>>>(score, attnb);

    // 6) out[h] = attn[h] · Vs[h]  via NT GEMM against pre-transposed vtb
    {
        const dim3 g(CHUNK / BM, D_MODEL / BN, N_HEADS);  // (1, 32, 16)
        gemm_nt_wmma<true, false><<<g, blk, 0, stream>>>(
            attnb, vtb, nullptr, catb, CHUNK, D_MODEL, CHUNK,
            (long long)CHUNK * CHUNK, (long long)D_MODEL * CHUNK,
            (long long)CHUNK * D_MODEL);
    }

    // 7) final: out = concat · Wo^T + bo  (fp32 out)
    {
        const dim3 g(SEQ / BM, D_MODEL / BN, 1);
        gemm_nt_wmma<false, true><<<g, blk, 0, stream>>>(
            catb, Wob, bo, out, SEQ, D_MODEL, D_MODEL, 0, 0, 0);
    }
}
